// CrossAttention_47321949667810
// MI455X (gfx1250) — matrix operand order repaired and checked by reading, not compiled, not run on hardware
//
#include <hip/hip_runtime.h>

// ---------------- problem constants ----------------
#define SY   2048   // query sequence (y)
#define SX   2048   // key/value sequence (x)
#define DIMI 1024
#define NH   16
#define PD   64     // head dim
#define BB   2
#define MROWS (BB * SY)   // 4096

typedef __attribute__((ext_vector_type(16))) __bf16 v16bf;
typedef __attribute__((ext_vector_type(8)))  float  v8f;
typedef unsigned short u16b;

typedef unsigned int u32x4 __attribute__((ext_vector_type(4)));
typedef int          i32x4 __attribute__((ext_vector_type(4)));
typedef int          i32x8 __attribute__((ext_vector_type(8)));

#if defined(__has_builtin)
#if __has_builtin(__builtin_amdgcn_tensor_load_to_lds)
#define USE_TDM 1
#endif
#endif
#ifndef USE_TDM
#define USE_TDM 0
#endif

union FragU { uint4 q[2]; v16bf v; };

// ---------------- helpers ----------------
__device__ __forceinline__ v8f wmma_bf16(v16bf a, v16bf b, v8f c) {
  // 8 args: (neg_a, A, neg_b, B, c_mod, C, reuse_a, reuse_b)
  return __builtin_amdgcn_wmma_f32_16x16x32_bf16(
      false, a, false, b, (short)0, c, false, false);
}

__device__ __forceinline__ u16b f2bf(float f) {
  unsigned int u = __float_as_uint(f);
  u += 0x7fffu + ((u >> 16) & 1u);   // round-to-nearest-even
  return (u16b)(u >> 16);
}

// A-operand fragment (16x32 bf16): lane L holds row L&15.
__device__ __forceinline__ v16bf load_frag_a(const u16b* rowp, int k0, int hi) {
  FragU u;
  u.q[0] = *(const uint4*)(rowp + k0 + hi * 8);
  u.q[1] = *(const uint4*)(rowp + k0 + 16 + hi * 8);
  return u.v;
}

// B-operand fragment (32x16 bf16): lane holds one N column; K in table order (8h+s, 16+8h+s).
__device__ __forceinline__ v16bf load_frag_b(const u16b* rowp, int k0, int hi) {
  FragU u;
  u.q[0] = *(const uint4*)(rowp + k0 + hi * 8);
  u.q[1] = *(const uint4*)(rowp + k0 + 16 + hi * 8);
  return u.v;
}

#if USE_TDM
// Issue a 2D TDM load: tile_d1 rows x tile_d0 elems (bf16), row stride in elems.
// D# layout per CDNA5 ISA (group0 + group1, groups 2/3 zero).
// 6-arg builtin form: (g0, g1, g2, g3, g4, cpol).
__device__ __forceinline__ void tdm_load_2d(unsigned lds_addr, const void* gaddr,
                                            unsigned tile_d0, unsigned tile_d1,
                                            unsigned long long stride_elems) {
  unsigned long long ga = (unsigned long long)gaddr;
  u32x4 g0 = {
      0x1u,                                               // count=1, user mode
      lds_addr,                                           // lds_addr (bytes)
      (unsigned)ga,                                       // global_addr[31:0]
      (unsigned)((ga >> 32) & 0x01FFFFFFu) | 0x80000000u  // addr[56:32] | type=2
  };
  i32x8 g1 = {
      (int)0x00010000u,                                   // data_size=1 (2 bytes)
      (int)((tile_d0 & 0xffffu) << 16),                   // tensor_dim0[15:0] @63:48
      (int)((tile_d0 >> 16) | ((tile_d1 & 0xffffu) << 16)), // dim0 hi | tensor_dim1 lo
      (int)((tile_d1 >> 16) | ((tile_d0 & 0xffffu) << 16)), // dim1 hi | tile_dim0
      (int)(tile_d1 & 0xffffu),                           // tile_dim1 (tile_dim2=0)
      (int)(stride_elems & 0xffffffffu),                  // tensor_dim0_stride[31:0]
      (int)((stride_elems >> 32) & 0xffffu),              // stride[47:32]
      0                                                   // tensor_dim1_stride hi
  };
  i32x4 z4 = {0, 0, 0, 0};
  i32x8 z8 = {0, 0, 0, 0, 0, 0, 0, 0};
  __builtin_amdgcn_tensor_load_to_lds(g0, g1, z4, z4, z8, 0);
}
#endif

// ---------------- kernel 0: f32 -> bf16 convert ----------------
__global__ void cvt_bf16_kernel(const float* __restrict__ src,
                                u16b* __restrict__ dst, int n) {
  int i = blockIdx.x * blockDim.x + threadIdx.x;
  int stride = gridDim.x * blockDim.x;
  for (; i < n; i += stride) dst[i] = f2bf(src[i]);
}

// ---------------- kernel 1: q = y @ Wq^T + bq  ->  bf16 [B,H,SY,PD] ----------------
__global__ void __launch_bounds__(256)
qproj_kernel(const u16b* __restrict__ act,   // [MROWS][DIMI] bf16 (y)
             const u16b* __restrict__ w,     // [DIMI][DIMI]  bf16 (Wq, row = out ch)
             const float* __restrict__ bias, // [DIMI]
             u16b* __restrict__ qw) {        // [B,H,SY,PD]
  const int h    = blockIdx.x;               // 0..15 : head == 64-col block
  const int wave = threadIdx.x >> 5;
  const int lane = threadIdx.x & 31;
  const int lr = lane & 15, hi = lane >> 4;
  const int row0 = blockIdx.y * 128 + wave * 16;

  const u16b* arow = act + (size_t)(row0 + lr) * DIMI;
  v8f acc[4] = {};
  for (int k = 0; k < DIMI; k += 32) {
    v16bf af = load_frag_a(arow, k, hi);
#pragma unroll
    for (int nt = 0; nt < 4; ++nt) {
      const u16b* wrow = w + (size_t)(h * 64 + nt * 16 + lr) * DIMI;
      v16bf bf = load_frag_b(wrow, k, hi);
      acc[nt] = wmma_bf16(af, bf, acc[nt]);
    }
  }
#pragma unroll
  for (int nt = 0; nt < 4; ++nt) {
    float bv = bias[h * 64 + nt * 16 + lr];
#pragma unroll
    for (int r = 0; r < 8; ++r) {
      int row = row0 + r + 8 * hi;
      int b = row >> 11, dy = row & (SY - 1);
      qw[(((size_t)(b * NH + h)) * SY + dy) * PD + nt * 16 + lr] =
          f2bf(acc[nt][r] + bv);
    }
  }
}

// ---------------- kernel 2: kv = x @ Wkv^T + bkv ----------------
// cols 0..1023 -> K bf16 [B,H,SX,PD]; cols 1024..2047 -> V bf16 TRANSPOSED [B,H,PD,SX]
__global__ void __launch_bounds__(256)
kvproj_kernel(const u16b* __restrict__ act,   // [MROWS][DIMI] bf16 (x)
              const u16b* __restrict__ w,     // [2*DIMI][DIMI] bf16 (Wkv)
              const float* __restrict__ bias, // [2*DIMI]
              u16b* __restrict__ kw,          // [B,H,SX,PD]
              u16b* __restrict__ vw) {        // [B,H,PD,SX]
  const int cb   = blockIdx.x;               // 0..31 : 64-col block of 2048 cols
  const int isV  = cb >> 4;                  // 0 = K, 1 = V
  const int h    = cb & 15;
  const int wave = threadIdx.x >> 5;
  const int lane = threadIdx.x & 31;
  const int lr = lane & 15, hi = lane >> 4;
  const int row0 = blockIdx.y * 128 + wave * 16;

  const u16b* arow = act + (size_t)(row0 + lr) * DIMI;
  v8f acc[4] = {};
  for (int k = 0; k < DIMI; k += 32) {
    v16bf af = load_frag_a(arow, k, hi);
#pragma unroll
    for (int nt = 0; nt < 4; ++nt) {
      const u16b* wrow = w + (size_t)(cb * 64 + nt * 16 + lr) * DIMI;
      v16bf bf = load_frag_b(wrow, k, hi);
      acc[nt] = wmma_bf16(af, bf, acc[nt]);
    }
  }
#pragma unroll
  for (int nt = 0; nt < 4; ++nt) {
    float bv = bias[cb * 64 + nt * 16 + lr];
#pragma unroll
    for (int r = 0; r < 8; ++r) {
      int row = row0 + r + 8 * hi;
      int b = row >> 11, dx = row & (SX - 1);
      int p = nt * 16 + lr;
      u16b val = f2bf(acc[nt][r] + bv);
      if (isV)
        vw[(((size_t)(b * NH + h)) * PD + p) * SX + dx] = val;
      else
        kw[(((size_t)(b * NH + h)) * SX + dx) * PD + p] = val;
    }
  }
}

// ---------------- kernel 3: flash attention ----------------
// grid.x = B*H*(SY/128); 8 waves/block, each wave owns 16 query rows.
// K/V tiles (64 keys x 64 headdim) staged into LDS by the Tensor Data Mover,
// double-buffered, tracked with TENSORcnt; published to all waves by barriers.
__global__ void __launch_bounds__(256)
attn_kernel(const u16b* __restrict__ qw,  // [B,H,SY,PD]
            const u16b* __restrict__ kw,  // [B,H,SX,PD]
            const u16b* __restrict__ vw,  // [B,H,PD,SX]
            float* __restrict__ out) {    // [B,H,SY,PD] flat (raw reshape)
  const int bh    = blockIdx.x >> 4;      // 0..31
  const int qtile = blockIdx.x & 15;
  const int wave  = threadIdx.x >> 5;
  const int lane  = threadIdx.x & 31;
  const int lr = lane & 15, hi = lane >> 4;
  const int row0 = qtile * 128 + wave * 16;

  const u16b* Q = qw + (size_t)bh * SY * PD;
  const u16b* K = kw + (size_t)bh * SX * PD;
  const u16b* V = vw + (size_t)bh * PD * SX;

  __shared__ u16b pbuf[8][16 * 64];       // per-wave P staging (row-major 16x64)
  u16b* pw = pbuf[wave];
#if USE_TDM
  __shared__ u16b kbuf[2][64][64];        // [key][headdim], double buffered
  __shared__ u16b vbuf[2][64][64];        // [headdim][key], double buffered
#endif

  // Q fragments stay resident: 16 rows x 64 head dim = 2 A-fragments
  v16bf qf[2];
  {
    const u16b* qrow = Q + (size_t)(row0 + lr) * PD;
    qf[0] = load_frag_a(qrow, 0, hi);
    qf[1] = load_frag_a(qrow, 32, hi);
  }

  v8f O[4] = {};                          // 16 rows x 64 head-dim accumulator
  float mrow[8], lrow[8];
#pragma unroll
  for (int r = 0; r < 8; ++r) { mrow[r] = -1e30f; lrow[r] = 0.f; }

#if USE_TDM
  if (wave == 0) {                        // prime the pipeline: tile 0 -> buffer 0
    tdm_load_2d((unsigned)(unsigned long long)(void*)&kbuf[0][0][0],
                K, PD, 64, PD);
    tdm_load_2d((unsigned)(unsigned long long)(void*)&vbuf[0][0][0],
                V, 64, PD, SX);
  }
#endif

  for (int kb = 0; kb < SX; kb += 64) {
#if USE_TDM
    const int buf = (kb >> 6) & 1;
    if (wave == 0) {
      if (kb + 64 < SX) {                 // issue next tile, retire current (<=2 left)
        tdm_load_2d((unsigned)(unsigned long long)(void*)&kbuf[buf ^ 1][0][0],
                    K + (size_t)(kb + 64) * PD, PD, 64, PD);
        tdm_load_2d((unsigned)(unsigned long long)(void*)&vbuf[buf ^ 1][0][0],
                    V + (size_t)(kb + 64), 64, PD, SX);
        __builtin_amdgcn_s_wait_tensorcnt(2);
      } else {
        __builtin_amdgcn_s_wait_tensorcnt(0);
      }
    }
    __syncthreads();                      // tile kb visible to all waves
    const u16b* kbase = &kbuf[buf][0][0];
    const u16b* vbase = &vbuf[buf][0][0];
#else
    if (kb + 64 < SX) {                   // speculative prefetch of next K/V block
      __builtin_prefetch(K + (size_t)(kb + 64 + lr) * PD, 0, 1);
      __builtin_prefetch(V + (size_t)lr * SX + kb + 64, 0, 1);
    }
#endif
    // ---- S = Q K^T  (no 1/sqrt(d) scale, matching reference) ----
    v8f S[4] = {};
#pragma unroll
    for (int nt = 0; nt < 4; ++nt) {
#if USE_TDM
      const u16b* krow = kbase + (nt * 16 + lr) * PD;
#else
      const u16b* krow = K + (size_t)(kb + nt * 16 + lr) * PD;
#endif
      v16bf b0 = load_frag_b(krow, 0, hi);
      v16bf b1 = load_frag_b(krow, 32, hi);
      S[nt] = wmma_bf16(qf[0], b0, S[nt]);
      S[nt] = wmma_bf16(qf[1], b1, S[nt]);
    }
    // ---- online softmax (per row = VGPR index + lane half) ----
#pragma unroll
    for (int r = 0; r < 8; ++r) {
      float t = fmaxf(fmaxf(S[0][r], S[1][r]), fmaxf(S[2][r], S[3][r]));
      t = fmaxf(t, __shfl_xor(t, 1, 32));
      t = fmaxf(t, __shfl_xor(t, 2, 32));
      t = fmaxf(t, __shfl_xor(t, 4, 32));
      t = fmaxf(t, __shfl_xor(t, 8, 32));   // stays inside 16-lane half-group
      float mnew  = fmaxf(mrow[r], t);
      float scale = __expf(mrow[r] - mnew);
      mrow[r] = mnew;
      lrow[r] *= scale;
      float ps = 0.f;
#pragma unroll
      for (int nt = 0; nt < 4; ++nt) {
        float p = __expf(S[nt][r] - mnew);
        S[nt][r] = p;
        ps += p;
        O[nt][r] *= scale;
      }
      ps += __shfl_xor(ps, 1, 32);
      ps += __shfl_xor(ps, 2, 32);
      ps += __shfl_xor(ps, 4, 32);
      ps += __shfl_xor(ps, 8, 32);
      lrow[r] += ps;
      // C-layout -> row-major LDS tile (bf16)
#pragma unroll
      for (int nt = 0; nt < 4; ++nt)
        pw[(r + 8 * hi) * 64 + nt * 16 + lr] = f2bf(S[nt][r]);
    }
    asm volatile("s_wait_dscnt 0" ::: "memory");  // P tile visible (in-wave DS order)
    // reload P as A-operand fragments (transpose through LDS)
    const u16b* prow = pw + lr * 64;
    v16bf pf0 = load_frag_a(prow, 0, hi);
    v16bf pf1 = load_frag_a(prow, 32, hi);
    // ---- O += P V ----
#pragma unroll
    for (int nt = 0; nt < 4; ++nt) {
#if USE_TDM
      const u16b* vrow = vbase + (nt * 16 + lr) * 64;
#else
      const u16b* vrow = V + (size_t)(nt * 16 + lr) * SX + kb;
#endif
      v16bf v0 = load_frag_b(vrow, 0, hi);
      v16bf v1 = load_frag_b(vrow, 32, hi);
      O[nt] = wmma_bf16(pf0, v0, O[nt]);
      O[nt] = wmma_bf16(pf1, v1, O[nt]);
    }
#if USE_TDM
    __syncthreads();                      // all waves done with buffer before reuse
#endif
  }
  // ---- normalize + store (raw [B,H,SY,PD] flat == reference reshape) ----
  float inv[8];
#pragma unroll
  for (int r = 0; r < 8; ++r) inv[r] = 1.0f / lrow[r];
#pragma unroll
  for (int nt = 0; nt < 4; ++nt)
#pragma unroll
    for (int r = 0; r < 8; ++r) {
      int dy = row0 + r + 8 * hi;
      out[(((size_t)bh) * SY + dy) * PD + nt * 16 + lr] = O[nt][r] * inv[r];
    }
}

// ---------------- launch ----------------
extern "C" void kernel_launch(void* const* d_in, const int* in_sizes, int n_in,
                              void* d_out, int out_size, void* d_ws, size_t ws_size,
                              hipStream_t stream) {
  const float* x   = (const float*)d_in[0];   // [B,SX,DIM]
  const float* y   = (const float*)d_in[1];   // [B,SY,DIM]
  const float* Wq  = (const float*)d_in[2];   // [DIM,DIM]
  const float* bq  = (const float*)d_in[3];   // [DIM]
  const float* Wkv = (const float*)d_in[4];   // [2*DIM,DIM]
  const float* bkv = (const float*)d_in[5];   // [2*DIM]
  float* out = (float*)d_out;

  u16b* ws = (u16b*)d_ws;
  u16b* yb   = ws;                       // 4096*1024
  u16b* xb   = yb   + (size_t)MROWS * DIMI;
  u16b* wqb  = xb   + (size_t)MROWS * DIMI;
  u16b* wkvb = wqb  + (size_t)DIMI * DIMI;
  u16b* qw   = wkvb + (size_t)2 * DIMI * DIMI;
  u16b* kw   = qw   + (size_t)MROWS * DIMI;
  u16b* vw   = kw   + (size_t)MROWS * DIMI;

  cvt_bf16_kernel<<<2048, 256, 0, stream>>>(y,   yb,   MROWS * DIMI);
  cvt_bf16_kernel<<<2048, 256, 0, stream>>>(x,   xb,   MROWS * DIMI);
  cvt_bf16_kernel<<<1024, 256, 0, stream>>>(Wq,  wqb,  DIMI * DIMI);
  cvt_bf16_kernel<<<1024, 256, 0, stream>>>(Wkv, wkvb, 2 * DIMI * DIMI);

  qproj_kernel <<<dim3(16, MROWS / 128), 256, 0, stream>>>(yb, wqb,  bq,  qw);
  kvproj_kernel<<<dim3(32, MROWS / 128), 256, 0, stream>>>(xb, wkvb, bkv, kw, vw);

  attn_kernel<<<BB * NH * (SY / 128), 256, 0, stream>>>(qw, kw, vw, out);
}
